// MultiHeadAttention_87299505258698
// MI455X (gfx1250) — compile-verified, matrix operand order checked by reading, not run on hardware
//
#include <hip/hip_runtime.h>

typedef __attribute__((ext_vector_type(16))) _Float16 v16h;
typedef __attribute__((ext_vector_type(8)))  _Float16 v8h;
typedef __attribute__((ext_vector_type(8)))  float    v8f;

#define cB 2
#define cS 2048
#define cD 1024
#define cH 16
#define MT (cB * cS)          // 4096 tokens total

#define BM 128
#define BN 128
#define BKK 32
#define LDT 40                // padded LDS stride in halves (80B, 16B-multiple)

// ---------------------------------------------------------------------------
// WMMA helper: D = A(16x32 f16) * B(32x16 f16) + C(16x16 f32)
// ---------------------------------------------------------------------------
__device__ __forceinline__ v8f wmma16(v16h a, v16h b, v8f c) {
  return __builtin_amdgcn_wmma_f32_16x16x32_f16(
      false, a, false, b, (short)0, c, false, false);
}

// Load a 16x32 f16 A-fragment (or B-fragment given B^T row-major) from a
// row-major tile with leading dimension ld (in halves).
// Per ISA 7.12.2: lane = row + 16*g, halves[0:8]=K[8g..8g+7], halves[8:16]=K[16+8g..].
__device__ __forceinline__ v16h load_ab_frag(const _Float16* tile, int ld) {
  const int lane = threadIdx.x & 31;
  const int r  = lane & 15;
  const int ko = (lane >> 4) << 3;   // 0 or 8
  const _Float16* p = tile + (size_t)r * ld;
  v8h lo = *(const v8h*)(p + ko);
  v8h hi = *(const v8h*)(p + 16 + ko);
  v16h f;
#pragma unroll
  for (int i = 0; i < 8; ++i) { f[i] = lo[i]; f[i + 8] = hi[i]; }
  return f;
}

// ---------------------------------------------------------------------------
// Async global->LDS copy of a 128x32 f16 tile (8KB) into a padded LDS tile.
// 512 x 16B chunks, 256 threads x 2 chunks; each asm stmt is one ASYNCcnt op
// per wave (32 lanes x 16B). LDS address = low 32 bits of generic pointer.
// ---------------------------------------------------------------------------
__device__ __forceinline__ void async_fill_tile(const _Float16* __restrict__ g,
                                                int ld, _Float16* __restrict__ s) {
  const int t = threadIdx.x;
#pragma unroll
  for (int i = 0; i < 2; ++i) {
    const int c   = t + i * 256;       // chunk 0..511
    const int row = c >> 2;            // 0..127
    const int seg = (c & 3) << 3;      // 0,8,16,24 halves
    const uint64_t ga = (uint64_t)(uintptr_t)(g + (size_t)row * ld + seg);
    const uint32_t la = (uint32_t)(uintptr_t)(s + row * LDT + seg);
    asm volatile("global_load_async_to_lds_b128 %0, %1, off"
                 :: "v"(la), "v"(ga) : "memory");
  }
}

__device__ __forceinline__ void wait_async_zero() {
  asm volatile("s_wait_asynccnt 0" ::: "memory");
}

// ---------------------------------------------------------------------------
// 16-lane (DPP row) butterfly reductions: masks {1,2,7,15} generate all 16.
// ---------------------------------------------------------------------------
#if __has_builtin(__builtin_amdgcn_mov_dpp)
template <int CTRL>
__device__ __forceinline__ float dpp_mov_f32(float x) {
  return __int_as_float(
      __builtin_amdgcn_mov_dpp(__float_as_int(x), CTRL, 0xf, 0xf, true));
}
__device__ __forceinline__ float row16_max(float v) {
  v = fmaxf(v, dpp_mov_f32<0x0B1>(v));   // quad_perm [1,0,3,2]  (xor 1)
  v = fmaxf(v, dpp_mov_f32<0x04E>(v));   // quad_perm [2,3,0,1]  (xor 2)
  v = fmaxf(v, dpp_mov_f32<0x141>(v));   // row_half_mirror      (xor 7)
  v = fmaxf(v, dpp_mov_f32<0x140>(v));   // row_mirror           (xor 15)
  return v;
}
__device__ __forceinline__ float row16_sum(float v) {
  v += dpp_mov_f32<0x0B1>(v);
  v += dpp_mov_f32<0x04E>(v);
  v += dpp_mov_f32<0x141>(v);
  v += dpp_mov_f32<0x140>(v);
  return v;
}
#else
__device__ __forceinline__ float row16_max(float v) {
  for (int off = 8; off >= 1; off >>= 1) v = fmaxf(v, __shfl_xor(v, off, 32));
  return v;
}
__device__ __forceinline__ float row16_sum(float v) {
  for (int off = 8; off >= 1; off >>= 1) v += __shfl_xor(v, off, 32);
  return v;
}
#endif

// ---------------------------------------------------------------------------
// f32 -> f16 conversion pre-pass (8 elements per thread)
// ---------------------------------------------------------------------------
__global__ __launch_bounds__(256) void cvt_f32_f16(const float* __restrict__ src,
                                                   _Float16* __restrict__ dst,
                                                   int n) {
  const int i = (blockIdx.x * 256 + threadIdx.x) * 8;
  if (i + 8 <= n) {
    const float4* s4 = (const float4*)(src + i);
    float4 a = s4[0], b = s4[1];
    v8h o;
    o[0]=(_Float16)a.x; o[1]=(_Float16)a.y; o[2]=(_Float16)a.z; o[3]=(_Float16)a.w;
    o[4]=(_Float16)b.x; o[5]=(_Float16)b.y; o[6]=(_Float16)b.z; o[7]=(_Float16)b.w;
    *(v8h*)(dst + i) = o;
  }
}

// ---------------------------------------------------------------------------
// Y = X[M,Kd](f16) * W[N,Kd](f16)^T + bias[N](f32)
//   F32_OUT:   Y f32 [M,N];  TRANS_OUT: Y f16 [N,M];  else Y f16 [M,N]
// 128x128x32 block tile, 8 waves x (64x32) wave tile, double-buffered LDS
// filled by async global->LDS copies overlapped with WMMA.
// ---------------------------------------------------------------------------
template <int TRANS_OUT, int F32_OUT>
__global__ __launch_bounds__(256) void gemm_f16_xwt_bias(
    const _Float16* __restrict__ X, const _Float16* __restrict__ W,
    const float* __restrict__ bias, void* __restrict__ Y,
    int M, int N, int Kd) {
  __shared__ __align__(16) _Float16 As[2][BM * LDT];
  __shared__ __align__(16) _Float16 Bs[2][BN * LDT];

  const int m0   = blockIdx.y * BM;
  const int n0   = blockIdx.x * BN;
  const int wave = threadIdx.x >> 5;
  const int lane = threadIdx.x & 31;
  const int wm   = (wave >> 2) * 64;
  const int wn   = (wave & 3) * 32;

  v8f acc[4][2] = {};

  // Prologue: fill buffer 0
  async_fill_tile(X + (size_t)m0 * Kd, Kd, As[0]);
  async_fill_tile(W + (size_t)n0 * Kd, Kd, Bs[0]);
  wait_async_zero();
  __syncthreads();

  int cur = 0;
  for (int k0 = 0; k0 < Kd; k0 += BKK) {
    const int nxt = k0 + BKK;
    if (nxt < Kd) {   // overlap next tile's async copy with this tile's WMMAs
      async_fill_tile(X + (size_t)m0 * Kd + nxt, Kd, As[cur ^ 1]);
      async_fill_tile(W + (size_t)n0 * Kd + nxt, Kd, Bs[cur ^ 1]);
    }

    v16h a[4], b[2];
#pragma unroll
    for (int mi = 0; mi < 4; ++mi)
      a[mi] = load_ab_frag(&As[cur][(wm + mi * 16) * LDT], LDT);
#pragma unroll
    for (int ni = 0; ni < 2; ++ni)
      b[ni] = load_ab_frag(&Bs[cur][(wn + ni * 16) * LDT], LDT);
#pragma unroll
    for (int mi = 0; mi < 4; ++mi)
#pragma unroll
      for (int ni = 0; ni < 2; ++ni)
        acc[mi][ni] = wmma16(a[mi], b[ni], acc[mi][ni]);

    if (nxt < Kd) {
      wait_async_zero();
      __syncthreads();
      cur ^= 1;
    }
  }

  // Epilogue: C layout = lane(N + 16*(M>=8)), vgpr r = M mod 8.
  const int colr = lane & 15;
  const int rsel = (lane >> 4) << 3;
#pragma unroll
  for (int mi = 0; mi < 4; ++mi) {
#pragma unroll
    for (int ni = 0; ni < 2; ++ni) {
      const int gn = n0 + wn + ni * 16 + colr;
      const float bv = bias[gn];
#pragma unroll
      for (int r = 0; r < 8; ++r) {
        const int gm = m0 + wm + mi * 16 + r + rsel;
        const float v = acc[mi][ni][r] + bv;
        if (F32_OUT)
          ((float*)Y)[(size_t)gm * N + gn] = v;
        else if (TRANS_OUT)
          ((_Float16*)Y)[(size_t)gn * M + gm] = (_Float16)v;
        else
          ((_Float16*)Y)[(size_t)gm * N + gn] = (_Float16)v;
      }
    }
  }
}

// ---------------------------------------------------------------------------
// Flash-style attention: one wave per 16-query tile of one (b,h).
//   Q,Km: f16 [B*S, D];  Vt: f16 [D, B*S];  ctx: f16 [B*S, D] (merged heads)
// ---------------------------------------------------------------------------
__global__ __launch_bounds__(256) void attention_fa(
    const _Float16* __restrict__ Q, const _Float16* __restrict__ Km,
    const _Float16* __restrict__ Vt, const int* __restrict__ mask,
    _Float16* __restrict__ ctx) {
  __shared__ __align__(16) _Float16 Pl[8][16 * LDT];

  const int wave = threadIdx.x >> 5;
  const int lane = threadIdx.x & 31;
  const int bh = blockIdx.y;
  const int b  = bh >> 4;                 // H = 16
  const int h  = bh & 15;
  const int q0 = (blockIdx.x * 8 + wave) * 16;

  const int colr = lane & 15;
  const int prow = (lane >> 4) << 3;
  _Float16* pw = &Pl[wave][0];

  // Q fragments, pre-scaled by 1/sqrt(DK) = 0.125 (exact in f16)
  const _Float16* qbase = Q + (size_t)(b * cS + q0) * cD + h * 64;
  v16h aq0 = load_ab_frag(qbase, cD);
  v16h aq1 = load_ab_frag(qbase + 32, cD);
#pragma unroll
  for (int i = 0; i < 16; ++i) {
    aq0[i] = aq0[i] * (_Float16)0.125f;
    aq1[i] = aq1[i] * (_Float16)0.125f;
  }

  v8f   ctxa[4] = {};
  float mrow[8], lrow[8];
#pragma unroll
  for (int r = 0; r < 8; ++r) { mrow[r] = -3.0e38f; lrow[r] = 0.0f; }

  for (int k0 = 0; k0 < cS; k0 += 32) {
    const _Float16* kb = Km + (size_t)(b * cS + k0) * cD + h * 64;
    if (k0 + 32 < cS)  // gfx1250 global_prefetch_b8 of next K tile
      __builtin_prefetch(kb + (size_t)32 * cD, 0, 3);

    // ---- scores: 2 key sub-tiles x 2 dk chunks ----
    v16h bk00 = load_ab_frag(kb, cD);
    v16h bk01 = load_ab_frag(kb + 32, cD);
    v16h bk10 = load_ab_frag(kb + (size_t)16 * cD, cD);
    v16h bk11 = load_ab_frag(kb + (size_t)16 * cD + 32, cD);
    v8f s0 = {}, s1 = {};
    s0 = wmma16(aq0, bk00, s0); s0 = wmma16(aq1, bk01, s0);
    s1 = wmma16(aq0, bk10, s1); s1 = wmma16(aq1, bk11, s1);

    // ---- mask (key column = lane low bits) ----
    const int mk0 = mask[b * cS + k0 + colr];
    const int mk1 = mask[b * cS + k0 + 16 + colr];
#pragma unroll
    for (int r = 0; r < 8; ++r) {
      if (mk0 == 0) s0[r] = -1.0e9f;
      if (mk1 == 0) s1[r] = -1.0e9f;
    }

    // ---- online softmax: DPP row reductions inside 16-lane halves ----
    float corr[8];
#pragma unroll
    for (int r = 0; r < 8; ++r) {
      const float vmax = row16_max(fmaxf(s0[r], s1[r]));
      const float mn = fmaxf(mrow[r], vmax);
      const float c  = __expf(mrow[r] - mn);
      const float p0 = __expf(s0[r] - mn);
      const float p1 = __expf(s1[r] - mn);
      lrow[r] = lrow[r] * c + row16_sum(p0 + p1);
      mrow[r] = mn;
      corr[r] = c;
      s0[r] = p0; s1[r] = p1;
    }
#pragma unroll
    for (int d = 0; d < 4; ++d)
#pragma unroll
      for (int r = 0; r < 8; ++r) ctxa[d][r] *= corr[r];

    // ---- transpose P (C layout -> A-fragment layout) via LDS ----
    __syncthreads();   // uniform across block: protect previous iter's reads
#pragma unroll
    for (int r = 0; r < 8; ++r) {
      pw[(r + prow) * LDT + colr]      = (_Float16)s0[r];
      pw[(r + prow) * LDT + 16 + colr] = (_Float16)s1[r];
    }
    __syncthreads();
    v16h ap = load_ab_frag(pw, LDT);

    // ---- ctx += P * V  (V^T row-major supplies B fragments) ----
#pragma unroll
    for (int d = 0; d < 4; ++d) {
      v16h bv = load_ab_frag(
          Vt + (size_t)(h * 64 + d * 16) * MT + (size_t)b * cS + k0, MT);
      ctxa[d] = wmma16(ap, bv, ctxa[d]);
    }
  }

  // ---- normalize and store merged-head context ----
  float linv[8];
#pragma unroll
  for (int r = 0; r < 8; ++r) linv[r] = 1.0f / lrow[r];
#pragma unroll
  for (int d = 0; d < 4; ++d) {
#pragma unroll
    for (int r = 0; r < 8; ++r) {
      const size_t gm = (size_t)b * cS + q0 + r + prow;
      const int    gc = h * 64 + d * 16 + colr;
      ctx[gm * cD + gc] = (_Float16)(ctxa[d][r] * linv[r]);
    }
  }
}

// ---------------------------------------------------------------------------
// Host-side launcher
// ---------------------------------------------------------------------------
extern "C" void kernel_launch(void* const* d_in, const int* in_sizes, int n_in,
                              void* d_out, int out_size, void* d_ws, size_t ws_size,
                              hipStream_t stream) {
  (void)in_sizes; (void)n_in; (void)out_size; (void)ws_size;

  const float* q_in = (const float*)d_in[0];
  const float* k_in = (const float*)d_in[1];
  const float* v_in = (const float*)d_in[2];
  const int*   mask = (const int*)d_in[3];
  const float* w_q  = (const float*)d_in[4];
  const float* b_q  = (const float*)d_in[5];
  const float* w_k  = (const float*)d_in[6];
  const float* b_k  = (const float*)d_in[7];
  const float* w_v  = (const float*)d_in[8];
  const float* b_v  = (const float*)d_in[9];
  const float* w_o  = (const float*)d_in[10];
  const float* b_o  = (const float*)d_in[11];

  // Workspace layout (f16), total 32 Mi halves = 64 MiB
  _Float16* p   = (_Float16*)d_ws;
  _Float16* Xq  = p; p += (size_t)MT * cD;
  _Float16* Xk  = p; p += (size_t)MT * cD;
  _Float16* Xv  = p; p += (size_t)MT * cD;
  _Float16* Wqh = p; p += (size_t)cD * cD;
  _Float16* Wkh = p; p += (size_t)cD * cD;
  _Float16* Wvh = p; p += (size_t)cD * cD;
  _Float16* Woh = p; p += (size_t)cD * cD;
  _Float16* Qh  = p; p += (size_t)MT * cD;
  _Float16* Kh  = p; p += (size_t)MT * cD;
  _Float16* Vt  = p; p += (size_t)MT * cD;
  _Float16* Ch  = p;

  dim3 gblk(256);

  // f32 -> f16 pre-pass (activations + weights)
  const int nAct = MT * cD, nW = cD * cD;
  cvt_f32_f16<<<nAct / 2048, gblk, 0, stream>>>(q_in, Xq, nAct);
  cvt_f32_f16<<<nAct / 2048, gblk, 0, stream>>>(k_in, Xk, nAct);
  cvt_f32_f16<<<nAct / 2048, gblk, 0, stream>>>(v_in, Xv, nAct);
  cvt_f32_f16<<<nW   / 2048, gblk, 0, stream>>>(w_q, Wqh, nW);
  cvt_f32_f16<<<nW   / 2048, gblk, 0, stream>>>(w_k, Wkh, nW);
  cvt_f32_f16<<<nW   / 2048, gblk, 0, stream>>>(w_v, Wvh, nW);
  cvt_f32_f16<<<nW   / 2048, gblk, 0, stream>>>(w_o, Woh, nW);

  dim3 ggrid(cD / BN, MT / BM);                   // (8, 32)
  gemm_f16_xwt_bias<0, 0><<<ggrid, gblk, 0, stream>>>(Xq, Wqh, b_q, (void*)Qh, MT, cD, cD);
  gemm_f16_xwt_bias<0, 0><<<ggrid, gblk, 0, stream>>>(Xk, Wkh, b_k, (void*)Kh, MT, cD, cD);
  gemm_f16_xwt_bias<1, 0><<<ggrid, gblk, 0, stream>>>(Xv, Wvh, b_v, (void*)Vt, MT, cD, cD);

  dim3 agrid(cS / 16 / 8, cB * cH);               // (16, 32)
  attention_fa<<<agrid, gblk, 0, stream>>>(Qh, Kh, Vt, mask, Ch);

  gemm_f16_xwt_bias<0, 1><<<ggrid, gblk, 0, stream>>>(Ch, Woh, b_o, d_out, MT, cD, cD);
}
